// BatchTopKSAE_1589137900170
// MI455X (gfx1250) — matrix operand order repaired and checked by reading, not compiled, not run on hardware
//
#include <hip/hip_runtime.h>

// ---------------------------------------------------------------------------
// BatchTopK SAE forward for MI455X (gfx1250, wave32)
//   - all GEMMs: bf16 WMMA 16x16x32, f32 accum
//   - tiles staged via GLOBAL_LOAD_ASYNC_TO_LDS_B128 (ASYNCcnt), triple-buffered
//   - fp32->bf16 conversion hoisted into one-time prep kernels
// Sizes fixed by reference: B=4096, D=768, F=24576, K=64 -> NTOP=262144
// ---------------------------------------------------------------------------

#define B_ROWS   4096
#define D_DIM    768
#define F_DIM    24576
#define K_SEL    64
#define NTOP_SEL (K_SEL * B_ROWS)      /* 262144 */
#define TOPK_AUX 512
#define N_DEAD_T 5
#define AUX_PEN  0.03125f
#define L1_COEF  0.001f

typedef __attribute__((ext_vector_type(16))) __bf16 bf16x16;
typedef __attribute__((ext_vector_type(8)))  float  f32x8;
typedef int v4i __attribute__((vector_size(16)));   // matches builtin param type

union FragU { uint4 u[2]; bf16x16 f; };

#define AS1 __attribute__((address_space(1)))
#define AS3 __attribute__((address_space(3)))

#if __has_builtin(__builtin_amdgcn_global_load_async_to_lds_b128) && \
    __has_builtin(__builtin_amdgcn_s_wait_asynccnt)
#define USE_ASYNC 1
#else
#define USE_ASYNC 0
#endif

__device__ __forceinline__ unsigned short f2bf(float x) {
  unsigned int u = __float_as_uint(x);
  unsigned int r = u + 0x7FFFu + ((u >> 16) & 1u);   // round-to-nearest-even
  return (unsigned short)(r >> 16);
}

// ---------------------------------------------------------------------------
// bf16 WMMA GEMM: C[M,N] = A[M,K](bf16) @ BT[N,K](bf16)^T  (+bias, relu)
// Both A and BT are row-major with K contiguous -> identical 16B-chunk staging.
// Block 256 thr (8 waves), tile 128x128, BK=32, wave subtile 32x64.
// Triple-buffered LDS, async global->LDS staging, prefetch depth 2.
// ---------------------------------------------------------------------------
__launch_bounds__(256)
__global__ void wmma_gemm_bf16(const unsigned short* __restrict__ A,
                               const unsigned short* __restrict__ BT,
                               float* __restrict__ C, int ldc,
                               int Kdim,
                               const float* __restrict__ c_bias,
                               int relu)
{
  __shared__ __align__(16) unsigned short As[3][128 * 32];  // 3 x 8 KB
  __shared__ __align__(16) unsigned short Bs[3][128 * 32];  // 3 x 8 KB

  const int tid  = threadIdx.x;
  const int lane = tid & 31;
  const int wave = tid >> 5;
  const int wm   = wave & 3;          // 4 waves along M (32 rows)
  const int wn   = wave >> 2;         // 2 waves along N (64 cols)
  const int m0   = blockIdx.y * 128;
  const int n0   = blockIdx.x * 128;
  const int l15  = lane & 15;
  const int lhi  = (lane >= 16);

  // staging: 128 rows x 64 B = 512 chunks of 16 B; 2 chunks per thread per matrix
  const int r0 = tid >> 2;                   // rows 0..63
  const int r1 = (tid + 256) >> 2;           // rows 64..127
  const int s0 = (tid & 3) * 8;              // 8 bf16 per 16B segment

  auto stage = [&](int kt, int buf) {
    const unsigned short* gA0 = A  + (size_t)(m0 + r0) * Kdim + kt * 32 + s0;
    const unsigned short* gA1 = A  + (size_t)(m0 + r1) * Kdim + kt * 32 + s0;
    const unsigned short* gB0 = BT + (size_t)(n0 + r0) * Kdim + kt * 32 + s0;
    const unsigned short* gB1 = BT + (size_t)(n0 + r1) * Kdim + kt * 32 + s0;
#if USE_ASYNC
    __builtin_amdgcn_global_load_async_to_lds_b128(
        (AS1 v4i*)gA0, (AS3 v4i*)&As[buf][r0 * 32 + s0], 0, 0);
    __builtin_amdgcn_global_load_async_to_lds_b128(
        (AS1 v4i*)gA1, (AS3 v4i*)&As[buf][r1 * 32 + s0], 0, 0);
    __builtin_amdgcn_global_load_async_to_lds_b128(
        (AS1 v4i*)gB0, (AS3 v4i*)&Bs[buf][r0 * 32 + s0], 0, 0);
    __builtin_amdgcn_global_load_async_to_lds_b128(
        (AS1 v4i*)gB1, (AS3 v4i*)&Bs[buf][r1 * 32 + s0], 0, 0);
#else
    *(uint4*)&As[buf][r0 * 32 + s0] = *(const uint4*)gA0;
    *(uint4*)&As[buf][r1 * 32 + s0] = *(const uint4*)gA1;
    *(uint4*)&Bs[buf][r0 * 32 + s0] = *(const uint4*)gB0;
    *(uint4*)&Bs[buf][r1 * 32 + s0] = *(const uint4*)gB1;
#endif
  };

  f32x8 acc[2][4];
#pragma unroll
  for (int i = 0; i < 2; ++i)
#pragma unroll
    for (int j = 0; j < 4; ++j)
#pragma unroll
      for (int e = 0; e < 8; ++e) acc[i][j][e] = 0.0f;

  const int T = Kdim / 32;
  stage(0, 0);
  if (T > 1) stage(1, 1);

  for (int t = 0; t < T; ++t) {
    const int cur = t % 3;
#if USE_ASYNC
    // retire oldest async batch (tile t); tile t+1's 4 ops may stay in flight
    if (t + 1 < T) __builtin_amdgcn_s_wait_asynccnt(4);
    else           __builtin_amdgcn_s_wait_asynccnt(0);
#endif
    __syncthreads();              // everyone's tile-t data is in LDS now
    if (t + 2 < T) stage(t + 2, (t + 2) % 3);   // reuse distance 3: WAR-safe

    const unsigned short* as = As[cur];
    const unsigned short* bs = Bs[cur];
    FragU afr[2], bfr[4];
    // A 16-bit 16x32 layout: lanes0-15: K{0..7,16..23}; lanes16-31: K{8..15,24..31}
#pragma unroll
    for (int mt = 0; mt < 2; ++mt) {
      int r = wm * 32 + mt * 16 + l15;
      afr[mt].u[0] = *(const uint4*)&as[r * 32 + (lhi ? 8  : 0)];
      afr[mt].u[1] = *(const uint4*)&as[r * 32 + (lhi ? 24 : 16)];
    }
    // B 16-bit 32x16 layout: lanes0-15: K{0..7,16..23}; lanes16-31: K{8..15,24..31}; N = lane&15
#pragma unroll
    for (int nt = 0; nt < 4; ++nt) {
      int n = wn * 64 + nt * 16 + l15;
      bfr[nt].u[0] = *(const uint4*)&bs[n * 32 + (lhi ? 8 : 0)];
      bfr[nt].u[1] = *(const uint4*)&bs[n * 32 + (lhi ? 24 : 16)];
    }
#pragma unroll
    for (int mt = 0; mt < 2; ++mt)
#pragma unroll
      for (int nt = 0; nt < 4; ++nt)
        acc[mt][nt] = __builtin_amdgcn_wmma_f32_16x16x32_bf16(
            false, afr[mt].f, false, bfr[nt].f, (short)0, acc[mt][nt],
            false, false);
  }

  // epilogue: C layout VGPR r -> M=r (lanes0-15) / M=8+r (lanes16-31)
#pragma unroll
  for (int mt = 0; mt < 2; ++mt)
#pragma unroll
    for (int nt = 0; nt < 4; ++nt) {
      int n = n0 + wn * 64 + nt * 16 + l15;
      float bias = c_bias ? c_bias[n] : 0.0f;
#pragma unroll
      for (int r = 0; r < 8; ++r) {
        int m = m0 + wm * 32 + mt * 16 + r + (lhi ? 8 : 0);
        float v = acc[mt][nt][r] + bias;
        if (relu) v = fmaxf(v, 0.0f);
        C[(size_t)m * ldc + n] = v;
      }
    }
}

// ---------------------------------------------------------------------------
// Prep: x_bf16 = bf16(x - b_dec)   (encoder A matrix)
// ---------------------------------------------------------------------------
__global__ void cvt_x_bf16(const float* __restrict__ x,
                           const float* __restrict__ b_dec,
                           unsigned short* __restrict__ out)
{
  size_t i = ((size_t)blockIdx.x * blockDim.x + threadIdx.x) * 4;
  if (i >= (size_t)B_ROWS * D_DIM) return;
  int k = (int)(i % D_DIM);
  float4 v = *(const float4*)(x + i);
  v.x -= b_dec[k]; v.y -= b_dec[k + 1]; v.z -= b_dec[k + 2]; v.w -= b_dec[k + 3];
  uint2 pk;
  pk.x = (unsigned)f2bf(v.x) | ((unsigned)f2bf(v.y) << 16);
  pk.y = (unsigned)f2bf(v.z) | ((unsigned)f2bf(v.w) << 16);
  *(uint2*)(out + i) = pk;
}

// ---------------------------------------------------------------------------
// Prep: out[C][R] (bf16) = transpose(in[R][C] fp32)  -> K-contiguous B^T
// ---------------------------------------------------------------------------
__global__ void transpose_cvt(const float* __restrict__ in,
                              unsigned short* __restrict__ out, int R, int C)
{
  __shared__ float tile[32][33];
  const int c0 = blockIdx.x * 32, rr0 = blockIdx.y * 32;
  const int tx = threadIdx.x, ty = threadIdx.y;      // block (32, 8)
#pragma unroll
  for (int j = 0; j < 32; j += 8)
    tile[ty + j][tx] = in[(size_t)(rr0 + ty + j) * C + c0 + tx];
  __syncthreads();
#pragma unroll
  for (int j = 0; j < 32; j += 8)
    out[(size_t)(c0 + ty + j) * R + rr0 + tx] = f2bf(tile[tx][ty + j]);
}

// ---------------------------------------------------------------------------
// Zero-init of histograms / accumulators (must run every launch).
// ---------------------------------------------------------------------------
__global__ void zero_kernel(unsigned int* h1, unsigned int* h2,
                            float* cs, unsigned int* scal)
{
  int i = blockIdx.x * 256 + threadIdx.x;
  if (i < 65536) h1[i] = 0u;
  if (i < 32768) h2[i] = 0u;
  if (i < F_DIM) cs[i] = 0.0f;
  if (i < 16)    scal[i] = 0u;
}

// ---------------------------------------------------------------------------
// Global top-K threshold: two-level histogram on positive-float bit patterns.
// ---------------------------------------------------------------------------
__global__ void hist1_kernel(const float* __restrict__ acts,
                             unsigned int* __restrict__ hist1)
{
  const size_t n = (size_t)B_ROWS * F_DIM;
  for (size_t i = (size_t)blockIdx.x * blockDim.x + threadIdx.x; i < n;
       i += (size_t)gridDim.x * blockDim.x) {
    float v = acts[i];
    if (v > 0.0f) atomicAdd(&hist1[__float_as_uint(v) >> 15], 1u);
  }
}

__global__ void hist2_kernel(const float* __restrict__ acts,
                             const unsigned int* __restrict__ scal,
                             unsigned int* __restrict__ hist2)
{
  const unsigned int bstar = scal[6];
  const size_t n = (size_t)B_ROWS * F_DIM;
  for (size_t i = (size_t)blockIdx.x * blockDim.x + threadIdx.x; i < n;
       i += (size_t)gridDim.x * blockDim.x) {
    float v = acts[i];
    if (v > 0.0f) {
      unsigned int b = __float_as_uint(v);
      if ((b >> 15) == bstar) atomicAdd(&hist2[b & 0x7FFFu], 1u);
    }
  }
}

// Single-block suffix scan from the top bin: boundary bin for `target`.
__global__ void scan_select(const unsigned int* __restrict__ hist, int nbins,
                            const unsigned int* __restrict__ target_ptr,
                            unsigned int target_const,
                            unsigned int* __restrict__ out_bin,
                            unsigned int* __restrict__ out_need)
{
  __shared__ unsigned int s[256];
  __shared__ unsigned int run;
  __shared__ int done;
  const int tid = threadIdx.x;
  const unsigned int target = target_ptr ? *target_ptr : target_const;
  if (tid == 0) { run = 0u; done = 0; *out_bin = 0u; *out_need = target; }
  __syncthreads();
  for (int base = nbins - 256; base >= 0; base -= 256) {
    if (done) break;
    s[tid] = hist[base + tid];
    __syncthreads();
    for (int off = 1; off < 256; off <<= 1) {          // suffix-inclusive scan
      unsigned int v = (tid + off < 256) ? s[tid + off] : 0u;
      __syncthreads();
      s[tid] += v;
      __syncthreads();
    }
    unsigned int incl = run + s[tid];
    unsigned int excl = run + ((tid < 255) ? s[tid + 1] : 0u);
    if (incl >= target && excl < target) {             // unique boundary thread
      *out_bin  = (unsigned int)(base + tid);
      *out_need = target - excl;
      done = 1;
    }
    __syncthreads();
    if (tid == 0) run += s[0];
    __syncthreads();
  }
}

// ---------------------------------------------------------------------------
// Apply global threshold -> acts_topk (fp32 out) + bf16 copy for decoder A,
// column sums, l1/l0 accumulation. Tile: 128 features x 64 rows.
// ---------------------------------------------------------------------------
__global__ void mask_colsum(const float* __restrict__ acts,
                            const unsigned int* __restrict__ scal,
                            float* __restrict__ acts_topk,
                            unsigned short* __restrict__ topk_bf,
                            float* __restrict__ colsum,
                            float* __restrict__ scalf)
{
  __shared__ float partial[256];
  __shared__ float red[256];
  const unsigned int thr_bits = (scal[6] << 15) | scal[8];
  const int f  = blockIdx.x * 128 + (threadIdx.x & 127);
  const int r0 = blockIdx.y * 64 + (threadIdx.x >> 7) * 32;
  float cs = 0.f, l1 = 0.f, l0 = 0.f;
  for (int r = 0; r < 32; ++r) {
    size_t idx = (size_t)(r0 + r) * F_DIM + f;
    float v = acts[idx];
    bool keep = (v > 0.0f) && (__float_as_uint(v) >= thr_bits);
    float o = keep ? v : 0.0f;
    acts_topk[idx] = o;
    topk_bf[idx] = keep ? f2bf(v) : (unsigned short)0;
    cs += o; l1 += o; l0 += keep ? 1.0f : 0.0f;
  }
  partial[threadIdx.x] = cs;
  __syncthreads();
  if (threadIdx.x < 128)
    atomicAdd(&colsum[f], partial[threadIdx.x] + partial[threadIdx.x + 128]);
  red[threadIdx.x] = l1; __syncthreads();
  for (int o = 128; o > 0; o >>= 1) {
    if ((int)threadIdx.x < o) red[threadIdx.x] += red[threadIdx.x + o];
    __syncthreads();
  }
  if (threadIdx.x == 0) atomicAdd(&scalf[0], red[0]);
  red[threadIdx.x] = l0; __syncthreads();
  for (int o = 128; o > 0; o >>= 1) {
    if ((int)threadIdx.x < o) red[threadIdx.x] += red[threadIdx.x + o];
    __syncthreads();
  }
  if (threadIdx.x == 0) atomicAdd(&scalf[1], red[0]);
}

// ---------------------------------------------------------------------------
// Dead-feature bookkeeping (does NOT mutate input nb array).
// ---------------------------------------------------------------------------
__global__ void dead_kernel(const float* __restrict__ colsum,
                            const int* __restrict__ nb_in,
                            unsigned char* __restrict__ dead,
                            unsigned int* __restrict__ scalu)
{
  int f = blockIdx.x * 256 + threadIdx.x;
  if (f >= F_DIM) return;
  float cs = colsum[f];
  int nb = (cs > 1e-6f) ? 0 : (nb_in[f] + ((cs < 1e-6f) ? 1 : 0));
  unsigned char d = (nb >= N_DEAD_T) ? 1u : 0u;
  dead[f] = d;
  if (d) atomicAdd(&scalu[4], 1u);                 // dead.any() count
  if (nb > N_DEAD_T) atomicAdd(&scalu[5], 1u);     // num_dead_features
}

// ---------------------------------------------------------------------------
// Sum of squared diff (+ optional residual store).
// ---------------------------------------------------------------------------
__global__ void sq_diff_kernel(const float* __restrict__ a,
                               const float* __restrict__ b,
                               float* __restrict__ resid,
                               float* __restrict__ accum, size_t n)
{
  __shared__ float red[256];
  float s = 0.0f;
  for (size_t i = (size_t)blockIdx.x * blockDim.x + threadIdx.x; i < n;
       i += (size_t)gridDim.x * blockDim.x) {
    float d = a[i] - b[i];
    if (resid) resid[i] = d;
    s += d * d;
  }
  red[threadIdx.x] = s; __syncthreads();
  for (int o = 128; o > 0; o >>= 1) {
    if ((int)threadIdx.x < o) red[threadIdx.x] += red[threadIdx.x + o];
    __syncthreads();
  }
  if (threadIdx.x == 0) atomicAdd(accum, red[0]);
}

// ---------------------------------------------------------------------------
// Per-row top-512 threshold over dead-masked acts (4096-bin LDS histogram).
// ---------------------------------------------------------------------------
__global__ void rowthr_kernel(const float* __restrict__ acts,
                              const unsigned char* __restrict__ dead,
                              unsigned int* __restrict__ row_thr)
{
  __shared__ unsigned int h[4096];
  __shared__ unsigned int tsum[256];
  const int row = blockIdx.x;
  const int tid = threadIdx.x;
  for (int i = tid; i < 4096; i += 256) h[i] = 0u;
  __syncthreads();
  const float* ar = acts + (size_t)row * F_DIM;
  for (int f = tid; f < F_DIM; f += 256) {
    float v = ar[f];
    if (v > 0.0f && dead[f]) atomicAdd(&h[__float_as_uint(v) >> 19], 1u);
  }
  __syncthreads();
  unsigned int sloc = 0;
  for (int j = 0; j < 16; ++j) sloc += h[tid * 16 + j];
  tsum[tid] = sloc;
  __syncthreads();
  if (tid == 0) {
    unsigned int run = 0, result = 0;
    bool found = false;
    for (int t = 255; t >= 0 && !found; --t) {
      if (run + tsum[t] >= (unsigned)TOPK_AUX) {
        for (int j = 15; j >= 0; --j) {
          unsigned int c = h[t * 16 + j];
          if (run + c >= (unsigned)TOPK_AUX) {
            result = (unsigned)(t * 16 + j) << 19; found = true; break;
          }
          run += c;
        }
        if (!found) { result = (unsigned)(t * 16) << 19; found = true; }
      } else {
        run += tsum[t];
      }
    }
    row_thr[row] = found ? result : 0u;   // <512 candidates -> take all
  }
}

// ---------------------------------------------------------------------------
// auxA_bf16[b][f] = bf16(acts) where dead[f] && v>0 && bits>=row_thr[b], else 0
// ---------------------------------------------------------------------------
__global__ void aux_mask_bf16(const float* __restrict__ acts,
                              const unsigned char* __restrict__ dead,
                              const unsigned int* __restrict__ row_thr,
                              unsigned short* __restrict__ out)
{
  size_t i = ((size_t)blockIdx.x * blockDim.x + threadIdx.x) * 4;
  if (i >= (size_t)B_ROWS * F_DIM) return;
  const int row = (int)(i / F_DIM);
  const int f   = (int)(i % F_DIM);
  const unsigned int tb = row_thr[row];
  float4 v = *(const float4*)(acts + i);
  unsigned short b0 = (dead[f]     && v.x > 0.f && __float_as_uint(v.x) >= tb) ? f2bf(v.x) : 0;
  unsigned short b1 = (dead[f + 1] && v.y > 0.f && __float_as_uint(v.y) >= tb) ? f2bf(v.y) : 0;
  unsigned short b2 = (dead[f + 2] && v.z > 0.f && __float_as_uint(v.z) >= tb) ? f2bf(v.z) : 0;
  unsigned short b3 = (dead[f + 3] && v.w > 0.f && __float_as_uint(v.w) >= tb) ? f2bf(v.w) : 0;
  uint2 pk;
  pk.x = (unsigned)b0 | ((unsigned)b1 << 16);
  pk.y = (unsigned)b2 | ((unsigned)b3 << 16);
  *(uint2*)(out + i) = pk;
}

// ---------------------------------------------------------------------------
// Final scalar outputs.
// ---------------------------------------------------------------------------
__global__ void finalize_kernel(const float* __restrict__ scalf,
                                const unsigned int* __restrict__ scalu,
                                float* __restrict__ out_scal)
{
  if (threadIdx.x == 0 && blockIdx.x == 0) {
    const float inv_bd = 1.0f / (float)(B_ROWS * D_DIM);
    float l2  = scalf[2] * inv_bd;
    float l1n = scalf[0] / (float)B_ROWS;
    float l0  = scalf[1] / (float)B_ROWS;
    float aux = (scalu[4] > 0u) ? (AUX_PEN * scalf[3] * inv_bd) : 0.0f;
    out_scal[0] = l2 + aux;               // loss
    out_scal[1] = l2;                     // l2_loss
    out_scal[2] = L1_COEF * l1n;          // l1_loss
    out_scal[3] = l0;                     // l0_norm
    out_scal[4] = l1n;                    // l1_norm
    out_scal[5] = aux;                    // aux_loss
    out_scal[6] = (float)scalu[5];        // num_dead_features
  }
}

// ---------------------------------------------------------------------------
extern "C" void kernel_launch(void* const* d_in, const int* in_sizes, int n_in,
                              void* d_out, int out_size, void* d_ws, size_t ws_size,
                              hipStream_t stream)
{
  const float* x     = (const float*)d_in[0];
  const float* W_enc = (const float*)d_in[1];
  const float* W_dec = (const float*)d_in[2];
  const float* b_dec = (const float*)d_in[3];
  const int*   nb_in = (const int*)d_in[4];

  float* out       = (float*)d_out;
  float* sae_out   = out;                                   // [B*D]
  float* acts_topk = out + (size_t)B_ROWS * D_DIM;          // [B*F]
  float* out_scal  = acts_topk + (size_t)B_ROWS * F_DIM;    // 7 scalars

  char* p = (char*)d_ws;
  float*          acts     = (float*)p;          p += (size_t)B_ROWS * F_DIM * 4;
  unsigned short* topk_bf  = (unsigned short*)p; p += (size_t)B_ROWS * F_DIM * 2;
  unsigned short* auxA_bf  = (unsigned short*)p; p += (size_t)B_ROWS * F_DIM * 2;
  unsigned short* x_bf     = (unsigned short*)p; p += (size_t)B_ROWS * D_DIM * 2;
  unsigned short* WencT_bf = (unsigned short*)p; p += (size_t)F_DIM * D_DIM * 2;
  unsigned short* WdecT_bf = (unsigned short*)p; p += (size_t)D_DIM * F_DIM * 2;
  float*          residual = (float*)p;          p += (size_t)B_ROWS * D_DIM * 4;
  float*          xr_aux   = (float*)p;          p += (size_t)B_ROWS * D_DIM * 4;
  float*          colsum   = (float*)p;          p += (size_t)F_DIM * 4;
  unsigned int*   hist1    = (unsigned int*)p;   p += 65536 * 4;
  unsigned int*   hist2    = (unsigned int*)p;   p += 32768 * 4;
  unsigned int*   row_thr  = (unsigned int*)p;   p += (size_t)B_ROWS * 4;
  unsigned char*  dead     = (unsigned char*)p;  p += ((F_DIM + 3) / 4) * 4;
  float*          scalf    = (float*)p;
  unsigned int*   scalu    = (unsigned int*)p;   p += 16 * 4;

  // 0. zero accumulators + one-time bf16 prep
  zero_kernel<<<256, 256, 0, stream>>>(hist1, hist2, colsum, scalu);
  cvt_x_bf16<<<(B_ROWS * D_DIM / 4 + 255) / 256, 256, 0, stream>>>(x, b_dec, x_bf);
  transpose_cvt<<<dim3(F_DIM / 32, D_DIM / 32), dim3(32, 8), 0, stream>>>(
      W_enc, WencT_bf, D_DIM, F_DIM);                       // -> [F][D]
  transpose_cvt<<<dim3(D_DIM / 32, F_DIM / 32), dim3(32, 8), 0, stream>>>(
      W_dec, WdecT_bf, F_DIM, D_DIM);                       // -> [D][F]

  // 1. encoder: acts = relu((x - b_dec) @ W_enc)
  wmma_gemm_bf16<<<dim3(F_DIM / 128, B_ROWS / 128), 256, 0, stream>>>(
      x_bf, WencT_bf, acts, F_DIM, D_DIM, nullptr, 1);

  // 2. global batch top-K threshold (two-level bit-pattern histogram)
  hist1_kernel<<<2048, 256, 0, stream>>>(acts, hist1);
  scan_select<<<1, 256, 0, stream>>>(hist1, 65536, nullptr, (unsigned)NTOP_SEL,
                                     scalu + 6, scalu + 7);
  hist2_kernel<<<2048, 256, 0, stream>>>(acts, scalu, hist2);
  scan_select<<<1, 256, 0, stream>>>(hist2, 32768, scalu + 7, 0u,
                                     scalu + 8, scalu + 9);

  // 3. scatter-mask -> acts_topk (+bf16 copy); col sums + l1/l0
  mask_colsum<<<dim3(F_DIM / 128, B_ROWS / 64), 256, 0, stream>>>(
      acts, scalu, acts_topk, topk_bf, colsum, scalf);

  // 4. inactive-feature update / dead mask
  dead_kernel<<<(F_DIM + 255) / 256, 256, 0, stream>>>(colsum, nb_in, dead, scalu);

  // 5. decoder: sae_out = acts_topk @ W_dec + b_dec
  wmma_gemm_bf16<<<dim3(D_DIM / 128, B_ROWS / 128), 256, 0, stream>>>(
      topk_bf, WdecT_bf, sae_out, D_DIM, F_DIM, b_dec, 0);

  // 6. l2 loss + residual
  sq_diff_kernel<<<1024, 256, 0, stream>>>(x, sae_out, residual, scalf + 2,
                                           (size_t)B_ROWS * D_DIM);

  // 7. per-row aux top-512 thresholds + masked bf16 A matrix
  rowthr_kernel<<<B_ROWS, 256, 0, stream>>>(acts, dead, row_thr);
  aux_mask_bf16<<<(int)((size_t)B_ROWS * F_DIM / 4 / 256), 256, 0, stream>>>(
      acts, dead, row_thr, auxA_bf);

  // 8. aux GEMM: xr_aux = masked(acts) @ W_dec
  wmma_gemm_bf16<<<dim3(D_DIM / 128, B_ROWS / 128), 256, 0, stream>>>(
      auxA_bf, WdecT_bf, xr_aux, D_DIM, F_DIM, nullptr, 0);

  // 9. aux loss
  sq_diff_kernel<<<1024, 256, 0, stream>>>(xr_aux, residual, nullptr, scalf + 3,
                                           (size_t)B_ROWS * D_DIM);

  // 10. scalars
  finalize_kernel<<<1, 32, 0, stream>>>(scalf, scalu, out_scal);
}
